// AttentionGCNl_20005957665496
// MI455X (gfx1250) — matrix operand order repaired and checked by reading, not compiled, not run on hardware
//
#include <hip/hip_runtime.h>
#include <stdint.h>

#define NEG_SLOPE 0.2f

typedef __attribute__((ext_vector_type(16))) __bf16 v16bf;
typedef __attribute__((ext_vector_type(8)))  float  v8f;

union FragBF { v16bf v; uint4 q[2]; };

// ---- monotone float<->uint mapping for atomicMax on floats --------------
__device__ __forceinline__ unsigned fkey(float f) {
  unsigned u = __float_as_uint(f);
  return (u & 0x80000000u) ? ~u : (u | 0x80000000u);
}
__device__ __forceinline__ float funkey(unsigned k) {
  unsigned u = (k & 0x80000000u) ? (k ^ 0x80000000u) : ~k;
  return __uint_as_float(u);
}

__device__ __forceinline__ uint16_t f32_to_bf16_rne(float f) {
  unsigned u = __float_as_uint(f);
  return (uint16_t)((u + 0x7FFFu + ((u >> 16) & 1u)) >> 16);
}

// ---- conversions ---------------------------------------------------------
__global__ void k_f32_to_bf16(const float* __restrict__ src,
                              uint16_t* __restrict__ dst, int n) {
  int i = blockIdx.x * blockDim.x + threadIdx.x;
  if (i < n) dst[i] = f32_to_bf16_rne(src[i]);
}

// W row-major [K][Nc] fp32 -> col-major [Nc][K] bf16 (so each lane's B frag
// is two contiguous 16-byte runs)
__global__ void k_convert_w_t(const float* __restrict__ W,
                              uint16_t* __restrict__ wt, int K, int Nc) {
  int i = blockIdx.x * blockDim.x + threadIdx.x;
  if (i < K * Nc) {
    int k = i / Nc, n = i % Nc;
    wt[n * K + k] = f32_to_bf16_rne(W[i]);
  }
}

// ---- bf16 WMMA GEMM + fused attention coefficients -----------------------
// C[M,128] = A[M,128] * B[128,128]; 256 threads = 8 waves; block handles
// 16 rows; wave w handles cols [16w,16w+16) == head w's 16 features, so it
// also produces a_s[m,w] and a_d[m,w] via a 16-lane shuffle reduction.
__global__ void __launch_bounds__(256)
k_gemm_att(const uint16_t* __restrict__ A,   // [M][128] row-major bf16
           const uint16_t* __restrict__ Bt,  // [128][128] col-major bf16
           float* __restrict__ C,
           const float* __restrict__ att_s,  // [8][16]
           const float* __restrict__ att_d,  // [8][16]
           float* __restrict__ as_,          // [M][8]
           float* __restrict__ ad_,          // [M][8]
           int M) {
  const int K = 128, Nc = 128;
  int wave = threadIdx.x >> 5;   // == head index
  int lane = threadIdx.x & 31;
  int m0   = blockIdx.x * 16;
  int n0   = wave * 16;
  int half = lane >> 4;   // 0: lanes 0-15, 1: lanes 16-31
  int l16  = lane & 15;

  int arowi = m0 + l16; if (arowi >= M) arowi = M - 1;  // tail-safe load
  const uint16_t* arow = A  + (size_t)arowi * K;
  const uint16_t* bcol = Bt + (size_t)(n0 + l16) * K;

  v8f acc = {};
#pragma unroll
  for (int k0 = 0; k0 < K; k0 += 32) {
    FragBF a, b;
    // A 16x32 bf16 layout: lane<16 -> K={0..7,16..23}, lane>=16 -> +8
    a.q[0] = *reinterpret_cast<const uint4*>(arow + k0 + half * 8);
    a.q[1] = *reinterpret_cast<const uint4*>(arow + k0 + 16 + half * 8);
    // B 32x16: same K order as A: lane<16 -> K={0..7,16..23}, lane>=16 -> +8
    b.q[0] = *reinterpret_cast<const uint4*>(bcol + k0 + half * 8);
    b.q[1] = *reinterpret_cast<const uint4*>(bcol + k0 + 16 + half * 8);
    acc = __builtin_amdgcn_wmma_f32_16x16x32_bf16(false, a.v, false, b.v,
                                                  (short)0, acc, false, false);
  }

  // ---- store C (C/D layout: VGPR r, lane half -> row m0+r+half*8) ----
  if (m0 + 16 <= M) {          // fast path: no per-row branches
#pragma unroll
    for (int r = 0; r < 8; ++r)
      C[(size_t)(m0 + r + half * 8) * Nc + n0 + l16] = acc[r];
  } else {
#pragma unroll
    for (int r = 0; r < 8; ++r) {
      int m = m0 + r + half * 8;
      if (m < M) C[(size_t)m * Nc + n0 + l16] = acc[r];
    }
  }

  // ---- fused attention coefficients for head `wave` ----
  float ws = att_s[wave * 16 + l16];
  float wd = att_d[wave * 16 + l16];
  float ss[8], dd[8];
#pragma unroll
  for (int r = 0; r < 8; ++r) {
    float s = acc[r] * ws;
    float d = acc[r] * wd;
#pragma unroll
    for (int off = 1; off < 16; off <<= 1) {   // reduce within 16-lane half
      s += __shfl_xor(s, off, 32);
      d += __shfl_xor(d, off, 32);
    }
    ss[r] = s; dd[r] = d;
  }
  if (l16 == 0) {
#pragma unroll
    for (int r = 0; r < 8; ++r) {
      int m = m0 + r + half * 8;
      if (m < M) { as_[m * 8 + wave] = ss[r]; ad_[m * 8 + wave] = dd[r]; }
    }
  }
}

// ---- init accumulators ---------------------------------------------------
__global__ void k_init(float* __restrict__ outb, unsigned* __restrict__ emaxk,
                       float* __restrict__ denom, int N) {
  int i = blockIdx.x * blockDim.x + threadIdx.x;
  if (i < N * 128) outb[i] = 0.f;
  if (i < N * 8) { emaxk[i] = 0u; denom[i] = 0.f; }  // key 0 == "-inf" sentinel
}

// ---- edge passes ---------------------------------------------------------
__device__ __forceinline__ void edge_ends(const int* __restrict__ ei, int E,
                                          int e, int& s, int& d) {
  if (e < E) { s = ei[e]; d = ei[E + e]; } else { s = e - E; d = s; }
}

__global__ void k_edge_max(const int* __restrict__ ei, int E, int Etot,
                           const float* __restrict__ as_, const float* __restrict__ ad_,
                           unsigned* __restrict__ emaxk) {
  int i = blockIdx.x * blockDim.x + threadIdx.x;
  if (i >= Etot * 8) return;
  int e = i >> 3, hh = i & 7, s, d;
  edge_ends(ei, E, e, s, d);
  float v = as_[s * 8 + hh] + ad_[d * 8 + hh];
  v = v > 0.f ? v : NEG_SLOPE * v;
  atomicMax(&emaxk[d * 8 + hh], fkey(v));
}

__global__ void k_edge_denom(const int* __restrict__ ei, int E, int Etot,
                             const float* __restrict__ as_, const float* __restrict__ ad_,
                             const unsigned* __restrict__ emaxk, float* __restrict__ denom) {
  int i = blockIdx.x * blockDim.x + threadIdx.x;
  if (i >= Etot * 8) return;
  int e = i >> 3, hh = i & 7, s, d;
  edge_ends(ei, E, e, s, d);
  int id = d * 8 + hh;
  float v = as_[s * 8 + hh] + ad_[id];
  v = v > 0.f ? v : NEG_SLOPE * v;
  atomicAdd(&denom[id], __expf(v - funkey(emaxk[id])));
}

// one wave32 per edge; lane handles 4 consecutive features (head = lane/4)
__global__ void k_edge_agg(const int* __restrict__ ei, int E, int Etot,
                           const float* __restrict__ as_, const float* __restrict__ ad_,
                           const unsigned* __restrict__ emaxk, const float* __restrict__ denom,
                           const float* __restrict__ h, float* __restrict__ outb) {
  int gid = blockIdx.x * blockDim.x + threadIdx.x;
  int e = gid >> 5;
  if (e >= Etot) return;
  int lane = gid & 31;
  int s, d; edge_ends(ei, E, e, s, d);
  int hh = lane >> 2;
  int f  = lane << 2;
  int id = d * 8 + hh;
  __builtin_prefetch(h + (size_t)s * 128 + f, 0, 0);  // global_prefetch_b8
  float v = as_[s * 8 + hh] + ad_[id];
  v = v > 0.f ? v : NEG_SLOPE * v;
  float alpha = __expf(v - funkey(emaxk[id])) / denom[id];
  float4 hv = *reinterpret_cast<const float4*>(h + (size_t)s * 128 + f);
  float* op = outb + (size_t)d * 128 + f;
  atomicAdd(op + 0, hv.x * alpha);
  atomicAdd(op + 1, hv.y * alpha);
  atomicAdd(op + 2, hv.z * alpha);
  atomicAdd(op + 3, hv.w * alpha);
}

// ---- bias + ELU; optionally emit bf16 for next layer's GEMM --------------
__global__ void k_bias_elu(float* __restrict__ outb, const float* __restrict__ bias,
                           uint16_t* __restrict__ xbf, int N, int writeBF) {
  int i = blockIdx.x * blockDim.x + threadIdx.x;
  if (i >= N * 128) return;
  float v = outb[i] + bias[i & 127];
  v = v > 0.f ? v : (__expf(v) - 1.f);
  if (writeBF) xbf[i] = f32_to_bf16_rne(v);
  else         outb[i] = v;
}

// ---- final MLP head: relu(h @ w1 + b1) @ w2 + b2 -> [N,1] ----------------
__global__ void __launch_bounds__(256)
k_mlp(const float* __restrict__ h, const float* __restrict__ w1,
      const float* __restrict__ b1, const float* __restrict__ w2,
      const float* __restrict__ b2, float* __restrict__ out, int N) {
  int gid = blockIdx.x * blockDim.x + threadIdx.x;
  int n = gid >> 5, lane = gid & 31;
  if (n >= N) return;
  const float* row = h + (size_t)n * 128;
  float acc[16];
#pragma unroll
  for (int j = 0; j < 16; ++j) acc[j] = 0.f;
#pragma unroll
  for (int t = 0; t < 4; ++t) {
    int k = lane + t * 32;
    float xv = row[k];
    const float* wr = w1 + k * 16;
#pragma unroll
    for (int j = 0; j < 16; ++j) acc[j] += xv * wr[j];
  }
  float r = 0.f;
#pragma unroll
  for (int j = 0; j < 16; ++j) {
    float sv = acc[j];
#pragma unroll
    for (int off = 16; off > 0; off >>= 1) sv += __shfl_xor(sv, off, 32);
    float z = sv + b1[j];
    z = z > 0.f ? z : 0.f;
    r += z * w2[j];
  }
  if (lane == 0) out[n] = r + b2[0];
}

// ---- host orchestration --------------------------------------------------
extern "C" void kernel_launch(void* const* d_in, const int* in_sizes, int n_in,
                              void* d_out, int out_size, void* d_ws, size_t ws_size,
                              hipStream_t stream) {
  const float* x   = (const float*)d_in[0];
  const int*   ei  = (const int*)  d_in[1];
  const float* W1  = (const float*)d_in[2];
  const float* at_s1 = (const float*)d_in[3];
  const float* at_d1 = (const float*)d_in[4];
  const float* b1  = (const float*)d_in[5];
  const float* W2  = (const float*)d_in[6];
  const float* at_s2 = (const float*)d_in[7];
  const float* at_d2 = (const float*)d_in[8];
  const float* b2  = (const float*)d_in[9];
  const float* l1w = (const float*)d_in[10];
  const float* l1b = (const float*)d_in[11];
  const float* l2w = (const float*)d_in[12];
  const float* l2b = (const float*)d_in[13];
  float* out = (float*)d_out;

  const int N = in_sizes[0] / 128;
  const int E = in_sizes[1] / 2;
  const int Etot = E + N;

  char* ws = (char*)d_ws;
  auto carve = [&](size_t bytes) {
    char* p = ws; ws += (bytes + 255) & ~(size_t)255; return p;
  };
  float*    hbuf  = (float*)   carve((size_t)N * 128 * 4);
  float*    outb  = (float*)   carve((size_t)N * 128 * 4);
  float*    asb   = (float*)   carve((size_t)N * 8 * 4);
  float*    adb   = (float*)   carve((size_t)N * 8 * 4);
  unsigned* emaxk = (unsigned*)carve((size_t)N * 8 * 4);
  float*    denom = (float*)   carve((size_t)N * 8 * 4);
  uint16_t* xbf   = (uint16_t*)carve((size_t)N * 128 * 2);
  uint16_t* wbf   = (uint16_t*)carve((size_t)128 * 128 * 2);

  const int B = 256;
  auto cdiv = [](long a, long b) { return (int)((a + b - 1) / b); };
  int gFeat  = cdiv((long)N * 128, B);
  int gEdgeH = cdiv((long)Etot * 8, B);
  int gEdgeW = cdiv((long)Etot * 32, B);
  int gGemm  = cdiv(N, 16);

  // ---------------- layer 1 ----------------
  k_f32_to_bf16<<<gFeat, B, 0, stream>>>(x, xbf, N * 128);
  k_convert_w_t<<<cdiv(128 * 128, B), B, 0, stream>>>(W1, wbf, 128, 128);
  k_gemm_att<<<gGemm, 256, 0, stream>>>(xbf, wbf, hbuf, at_s1, at_d1, asb, adb, N);
  k_init<<<gFeat, B, 0, stream>>>(outb, emaxk, denom, N);
  k_edge_max  <<<gEdgeH, B, 0, stream>>>(ei, E, Etot, asb, adb, emaxk);
  k_edge_denom<<<gEdgeH, B, 0, stream>>>(ei, E, Etot, asb, adb, emaxk, denom);
  k_edge_agg  <<<gEdgeW, B, 0, stream>>>(ei, E, Etot, asb, adb, emaxk, denom, hbuf, outb);
  k_bias_elu<<<gFeat, B, 0, stream>>>(outb, b1, xbf, N, 1);  // -> bf16 input of L2

  // ---------------- layer 2 ----------------
  k_convert_w_t<<<cdiv(128 * 128, B), B, 0, stream>>>(W2, wbf, 128, 128);
  k_gemm_att<<<gGemm, 256, 0, stream>>>(xbf, wbf, hbuf, at_s2, at_d2, asb, adb, N);
  k_init<<<gFeat, B, 0, stream>>>(outb, emaxk, denom, N);
  k_edge_max  <<<gEdgeH, B, 0, stream>>>(ei, E, Etot, asb, adb, emaxk);
  k_edge_denom<<<gEdgeH, B, 0, stream>>>(ei, E, Etot, asb, adb, emaxk, denom);
  k_edge_agg  <<<gEdgeW, B, 0, stream>>>(ei, E, Etot, asb, adb, emaxk, denom, hbuf, outb);
  k_bias_elu<<<gFeat, B, 0, stream>>>(outb, b2, xbf, N, 0);  // in-place f32

  // ---------------- MLP head ----------------
  k_mlp<<<cdiv((long)N * 32, B), B, 0, stream>>>(outb, l1w, l1b, l2w, l2b, out, N);
}
